// VSMLM_63471026700810
// MI455X (gfx1250) — matrix operand order repaired and checked by reading, not compiled, not run on hardware
//
#include <hip/hip_runtime.h>
#include <hip/hip_bf16.h>
#include <math.h>
#include <stdint.h>

// ---------------- model constants ----------------
constexpr int cV = 50277, cD = 512, cH = 8, cDH = 64, cDFF = 2048, cW = 8;
constexpr int cB = 2, cL = 2048, cBL = cB * cL;   // 4096 rows
constexpr int cNIT = 2;

typedef __attribute__((ext_vector_type(16))) _Float16 v16h;
typedef __attribute__((ext_vector_type(8)))  _Float16 v8h;
typedef __attribute__((ext_vector_type(8)))  float    v8f;
typedef __attribute__((ext_vector_type(4)))  unsigned int v4u;
typedef __attribute__((ext_vector_type(8)))  int     v8i;
typedef __attribute__((ext_vector_type(4)))  int     v4i;

#if defined(__has_builtin)
#  if __has_builtin(__builtin_amdgcn_tensor_load_to_lds) && __has_builtin(__builtin_amdgcn_s_wait_tensorcnt)
#    define USE_TDM 1
#  endif
#endif
#ifndef USE_TDM
#  define USE_TDM 0
#endif

__device__ __forceinline__ float gelu_tanh(float x) {
    float x3 = x * x * x;
    return 0.5f * x * (1.f + tanhf(0.7978845608028654f * (x + 0.044715f * x3)));
}
__device__ __forceinline__ float sigmoidf_(float x) { return 1.f / (1.f + __expf(-x)); }

// ---------------- f32 -> f16 convert ----------------
__global__ void k_cvt_f16(const float* __restrict__ src, _Float16* __restrict__ dst, size_t n) {
    size_t i = (size_t)blockIdx.x * blockDim.x + threadIdx.x;
    if (i < n) dst[i] = (_Float16)src[i];
}

// ---------------- copy fp32 ----------------
__global__ void k_copy_f32(const float* __restrict__ src, float* __restrict__ dst, int n) {
    int i = blockIdx.x * blockDim.x + threadIdx.x;
    if (i < n) dst[i] = src[i];
}

// ---------------- embedding: x = tok_emb[ids] + pos_emb[:L] ----------------
__global__ void k_embed(const int* __restrict__ ids, const float* __restrict__ tok,
                        const float* __restrict__ pos, float* __restrict__ x) {
    size_t i = (size_t)blockIdx.x * blockDim.x + threadIdx.x;
    if (i >= (size_t)cBL * cD) return;
    int d  = (int)(i & (cD - 1));
    int bl = (int)(i >> 9);
    int l  = bl & (cL - 1);
    x[i] = tok[(size_t)ids[bl] * cD + d] + pos[(size_t)l * cD + d];
}

// ---------------- LayerNorm (D=512), one wave per row, f16 output ----------------
__global__ void k_ln(const float* __restrict__ x, const float* __restrict__ g,
                     const float* __restrict__ b, _Float16* __restrict__ out, int rows) {
    int wid  = threadIdx.x >> 5;
    int lane = threadIdx.x & 31;
    int row  = blockIdx.x * (blockDim.x >> 5) + wid;
    if (row >= rows) return;
    const float* xr = x + (size_t)row * cD;
    float s = 0.f, s2 = 0.f;
    for (int t = lane; t < cD; t += 32) { float v = xr[t]; s += v; s2 += v * v; }
    for (int o = 16; o > 0; o >>= 1) { s += __shfl_xor(s, o, 32); s2 += __shfl_xor(s2, o, 32); }
    float m  = s * (1.f / cD);
    float vr = s2 * (1.f / cD) - m * m;
    float rs = rsqrtf(vr + 1e-5f);
    _Float16* orow = out + (size_t)row * cD;
    for (int t = lane; t < cD; t += 32)
        orow[t] = (_Float16)((xr[t] - m) * rs * g[t] + b[t]);
}

// ---------------- WMMA GEMM with TDM-staged B tiles ----------------------------
// C[M,N] = act(A[M,K] @ Bw[N,K]^T + bias) + addsrc.  A,Bw f16, K-contiguous.
// Block: 256 thr = 8 waves; block tile 128(M) x 64(N); per-wave 16x64 (4 accums).
// B k-chunks (64 rows x 64 halves) are DMA'd by the Tensor Data Mover into LDS
// (double-buffered, pad_amount=16B per 128B row -> 144B LDS row stride, OOB rows
// of the vocab edge zero-filled by TDM).  B fragments are read with explicit
// inline-asm ds_load_b128 (the TDM builtin takes descriptor values only, so the
// compiler cannot see the LDS writes and would otherwise fold the loads).
constexpr int KC    = 64;     // k-chunk (halves)
constexpr int ROWH  = 72;     // padded LDS row stride in halves (144 B)
constexpr int BUFB  = 64 * ROWH * 2;  // 9216 bytes per buffer

__global__ __launch_bounds__(256)
void k_gemm_wmma(const _Float16* __restrict__ A, const _Float16* __restrict__ Bw,
                 const float* __restrict__ bias, const float* __restrict__ addsrc,
                 float* __restrict__ C, int M, int N, int K, int act) {
    __shared__ __attribute__((aligned(32))) _Float16 sB[2 * 64 * ROWH];

    const int lane = threadIdx.x & 31;
    const int wave = threadIdx.x >> 5;                 // 0..7
    const int m0   = (blockIdx.y * 8 + wave) * 16;     // M always multiple of 128
    const int n0   = blockIdx.x * 64;
    const int half = lane >> 4;
    const int lr   = lane & 15;
    const int nch  = K / KC;

    v8f acc[4] = {v8f{}, v8f{}, v8f{}, v8f{}};
    const _Float16* Ap = A + (size_t)(m0 + lr) * K + half * 16;

#if USE_TDM
    // LDS byte offset of sB: generic LDS-aperture addresses carry the LDS
    // offset in addr[31:0] (ISA 10.2), so truncation yields the DS address.
    const unsigned sbase = (unsigned)(uintptr_t)(&sB[0]);

    // ---- TDM issue helper: chunk c -> LDS buffer (c&1) --------------------
    auto tdm_issue = [&](int c) {
        unsigned ldsOff = sbase + (unsigned)((c & 1) * BUFB);
        unsigned long long ga =
            (unsigned long long)(uintptr_t)(Bw + (size_t)n0 * K + (size_t)c * KC);
        v4u g0;
        g0[0] = 1u;                                    // count=1, user descriptor
        g0[1] = ldsOff;                                // lds_addr (bytes)
        g0[2] = (unsigned)(ga & 0xFFFFFFFFu);          // global_addr[31:0]
        g0[3] = (unsigned)((ga >> 32) & 0x01FFFFFFu)   // global_addr[56:32]
              | (2u << 30);                            // type=2 (image)
        unsigned td0 = (unsigned)(K - c * KC);         // remaining elems in row
        unsigned td1 = (unsigned)(N - n0);             // remaining rows (OOB->0)
        v8i g1;
        g1[0] = (int)((1u << 16)                       // data_size = 2 bytes
                    | (1u << 20)                       // pad_enable
                    | (4u << 22)                       // pad_interval: 32 dwords (128B)
                    | (3u << 25));                     // pad_amount: 4 dwords (16B)
        g1[1] = (int)((td0 & 0xFFFFu) << 16);          // atomic_addr=0 | tdim0 lo
        g1[2] = (int)((td0 >> 16) | ((td1 & 0xFFFFu) << 16));
        g1[3] = (int)((td1 >> 16) | ((unsigned)KC << 16));   // tile_dim0 = 64
        g1[4] = 64;                                    // tile_dim1=64, tile_dim2=0
        g1[5] = K;                                     // tensor_dim0_stride lo32
        g1[6] = 0;
        g1[7] = 0;
        v4i z4 = {0, 0, 0, 0};
        v8i z8 = {0, 0, 0, 0, 0, 0, 0, 0};
        // 6-arg toolchain variant: (g0, g1, g2, g3, extra, cpol)
        __builtin_amdgcn_tensor_load_to_lds(g0, g1, z4, z4, z8, 0);
    };
    if (wave == 0) {
        tdm_issue(0);
        if (nch > 1) { tdm_issue(1); __builtin_amdgcn_s_wait_tensorcnt(1); }
        else         {               __builtin_amdgcn_s_wait_tensorcnt(0); }
    }
    __syncthreads();
#endif

    for (int c = 0; c < nch; ++c) {
#if USE_TDM
        const unsigned bufb = sbase + (unsigned)((c & 1) * BUFB);
#else
        // fallback: cooperative global->LDS copy (single buffer)
        __syncthreads();
        for (int e = threadIdx.x; e < 64 * (KC / 2); e += 256) {   // dwords
            int row = e >> 5, cb = e & 31;
            int rn = n0 + row;
            unsigned val = 0;
            if (rn < N)
                val = *(const unsigned*)(Bw + (size_t)rn * K + (size_t)c * KC + cb * 2);
            *(unsigned*)(sB + row * ROWH + cb * 2) = val;
        }
        __syncthreads();
#endif
#pragma unroll
        for (int kk = 0; kk < KC; kk += 32) {
            const _Float16* Aq = Ap + c * KC + kk;
            v8h a0 = *(const v8h*)(Aq);
            v8h a1 = *(const v8h*)(Aq + 8);
            v16h af = __builtin_shufflevector(a0, a1, 0, 1, 2, 3, 4, 5, 6, 7,
                                                      8, 9, 10, 11, 12, 13, 14, 15);
            v8h b0[4], b1[4];
#if USE_TDM
            // fragment j: rows j*16+lr (row stride 144B, j block = 2304B),
            // columns half*16+kk (+16B for the second b128 of the v16h)
            unsigned baddr = bufb + (unsigned)(lr * (ROWH * 2) + half * 32 + kk * 2);
            asm volatile(
                "ds_load_b128 %0, %8 offset:0\n\t"
                "ds_load_b128 %1, %8 offset:16\n\t"
                "ds_load_b128 %2, %8 offset:2304\n\t"
                "ds_load_b128 %3, %8 offset:2320\n\t"
                "ds_load_b128 %4, %8 offset:4608\n\t"
                "ds_load_b128 %5, %8 offset:4624\n\t"
                "ds_load_b128 %6, %8 offset:6912\n\t"
                "ds_load_b128 %7, %8 offset:6928\n\t"
                "s_wait_dscnt 0x0"
                : "=v"(b0[0]), "=v"(b1[0]), "=v"(b0[1]), "=v"(b1[1]),
                  "=v"(b0[2]), "=v"(b1[2]), "=v"(b0[3]), "=v"(b1[3])
                : "v"(baddr));
#else
            const _Float16* Bk = sB + half * 16 + kk;
#pragma unroll
            for (int j = 0; j < 4; ++j) {
                b0[j] = *(const v8h*)(Bk + (j * 16 + lr) * ROWH);
                b1[j] = *(const v8h*)(Bk + (j * 16 + lr) * ROWH + 8);
            }
#endif
#pragma unroll
            for (int j = 0; j < 4; ++j) {
                v16h bf = __builtin_shufflevector(b0[j], b1[j], 0, 1, 2, 3, 4, 5, 6, 7,
                                                  8, 9, 10, 11, 12, 13, 14, 15);
                acc[j] = __builtin_amdgcn_wmma_f32_16x16x32_f16(
                    false, af, false, bf, (short)0, acc[j], false, false);
            }
        }
#if USE_TDM
        __syncthreads();                               // all done reading buf (c&1)
        if (wave == 0) {
            if (c + 2 < nch) { tdm_issue(c + 2); __builtin_amdgcn_s_wait_tensorcnt(1); }
            else             {                   __builtin_amdgcn_s_wait_tensorcnt(0); }
        }
        __syncthreads();                               // buf ((c+1)&1) ready
#endif
    }

#pragma unroll
    for (int j = 0; j < 4; ++j) {
#pragma unroll
        for (int r = 0; r < 8; ++r) {
            int row = m0 + r + half * 8;               // C layout: VGPR r -> M=r / r+8
            int col = n0 + j * 16 + lr;
            if (col < N) {
                float v = acc[j][r];
                if (bias)   v += bias[col];
                if (act)    v  = gelu_tanh(v);
                if (addsrc) v += addsrc[(size_t)row * N + col];
                C[(size_t)row * N + col] = v;
            }
        }
    }
}

// ---------------- sliding-window attention (W=8, per (b,l,h) thread) ----------
__global__ void k_swattn(const float* __restrict__ q, const float* __restrict__ k,
                         const float* __restrict__ v, float* __restrict__ o, int stride) {
    int t = blockIdx.x * blockDim.x + threadIdx.x;
    if (t >= cBL * cH) return;
    int h  = t & (cH - 1);
    int bl = t >> 3;
    int l  = bl & (cL - 1);
    int b  = bl >> 11;
    int base_bl = b * cL;
    const float* qp = q + (size_t)bl * cD + h * cDH;

    float s[cW], mx = -3.4e38f;
#pragma unroll
    for (int w = 0; w < cW; ++w) {
        int kl = l - w * stride;
        if (kl < 0) { s[w] = -1e30f; }
        else {
            const float* kp = k + (size_t)(base_bl + kl) * cD + h * cDH;
            float d = 0.f;
            for (int j = 0; j < cDH; ++j) d += qp[j] * kp[j];
            s[w] = d * 0.125f;                          // dh^-0.5
        }
        mx = fmaxf(mx, s[w]);
    }
    float p[cW], den = 0.f;
#pragma unroll
    for (int w = 0; w < cW; ++w) { p[w] = __expf(s[w] - mx); den += p[w]; }
    float inv = 1.f / den;

    float* op = o + (size_t)bl * cD + h * cDH;
    for (int j = 0; j < cDH; ++j) {
        float a = 0.f;
#pragma unroll
        for (int w = 0; w < cW; ++w) {
            int kl = l - w * stride;
            if (kl < 0) continue;
            a += p[w] * v[(size_t)(base_bl + kl) * cD + h * cDH + j];
        }
        op[j] = a * inv;
    }
}

// ---------------- register attention pieces ----------------
__global__ void k_regq(const float* __restrict__ wq, const float* __restrict__ reg,
                       float* __restrict__ qv) {
    int d = threadIdx.x;                                // 512 threads, 1 block
    float a = 0.f;
    const float* row = wq + (size_t)d * cD;
    for (int j = 0; j < cD; ++j) a += row[j] * reg[j];
    qv[d] = a;
}

__global__ void k_regscore(const float* __restrict__ qv, const float* __restrict__ kb,
                           float* __restrict__ sc) {
    int t = blockIdx.x * blockDim.x + threadIdx.x;
    if (t >= cBL) return;
    const float* kp = kb + (size_t)t * cD;
    float a = 0.f;
    for (int j = 0; j < cD; ++j) a += qv[j] * kp[j];
    sc[t] = a * 0.044194173824159216f;                  // D^-0.5
}

__global__ void k_regupd(const float* __restrict__ sc, const float* __restrict__ vb,
                         float* __restrict__ reg) {
    __shared__ float red[cD];
    __shared__ float mxs[cB], sms[cB];
    int tid = threadIdx.x;                              // 512 threads, 1 block
    for (int b = 0; b < cB; ++b) {
        float m = -3.4e38f;
        for (int l = tid; l < cL; l += cD) m = fmaxf(m, sc[b * cL + l]);
        red[tid] = m; __syncthreads();
        for (int o = 256; o > 0; o >>= 1) { if (tid < o) red[tid] = fmaxf(red[tid], red[tid + o]); __syncthreads(); }
        if (tid == 0) mxs[b] = red[0];
        __syncthreads();
        float s = 0.f;
        for (int l = tid; l < cL; l += cD) s += __expf(sc[b * cL + l] - mxs[b]);
        red[tid] = s; __syncthreads();
        for (int o = 256; o > 0; o >>= 1) { if (tid < o) red[tid] += red[tid + o]; __syncthreads(); }
        if (tid == 0) sms[b] = red[0];
        __syncthreads();
    }
    float acc = 0.f;
    for (int b = 0; b < cB; ++b) {
        float inv = 1.f / sms[b], m = mxs[b];
        for (int l = 0; l < cL; ++l)
            acc += __expf(sc[b * cL + l] - m) * inv * vb[(size_t)(b * cL + l) * cD + tid];
    }
    reg[tid] += acc * (1.f / cB);
}

// ---------------- column mean of delta: sm[d] = mean_r (y2 - x) ----------------
__global__ void k_colmean(const float* __restrict__ y2, const float* __restrict__ x,
                          float* __restrict__ sm) {
    int d = blockIdx.x * blockDim.x + threadIdx.x;
    if (d >= cD) return;
    float a = 0.f;
    for (int r = 0; r < cBL; ++r) a += y2[(size_t)r * cD + d] - x[(size_t)r * cD + d];
    sm[d] = a * (1.f / cBL);
}

// ---------------- gate + write-gate + register update (1 block, 512 thr) -------
__global__ void k_gate(const float* __restrict__ gw, const float* __restrict__ gb,
                       const float* __restrict__ wgw, const float* __restrict__ wgb,
                       const float* __restrict__ ww, const float* __restrict__ smv,
                       float* __restrict__ reg, float* __restrict__ gate) {
    __shared__ float red[cD];
    int d = threadIdx.x;
    const float* row = gw + (size_t)d * (2 * cD);
    float a = 0.f;
    for (int j = 0; j < cD; ++j) a += row[j] * reg[j];          // reg read (pre-update)
    for (int j = 0; j < cD; ++j) a += row[cD + j] * smv[j];
    gate[d] = sigmoidf_(a + gb[d]);
    red[d] = wgw[d] * smv[d];
    __syncthreads();
    for (int o = 256; o > 0; o >>= 1) { if (d < o) red[d] += red[d + o]; __syncthreads(); }
    float wg = sigmoidf_(red[0] + wgb[0]);
    float wd = 0.f;
    const float* wrow = ww + (size_t)d * cD;
    for (int j = 0; j < cD; ++j) wd += wrow[j] * smv[j];
    reg[d] += wg * wd;                                           // safe: reg reads done pre-sync
}

// ---------------- x = x + gate * (y2 - x) ----------------
__global__ void k_xupd(float* __restrict__ x, const float* __restrict__ y2,
                       const float* __restrict__ gate) {
    size_t i = (size_t)blockIdx.x * blockDim.x + threadIdx.x;
    if (i >= (size_t)cBL * cD) return;
    int d = (int)(i & (cD - 1));
    float xv = x[i];
    x[i] = xv + gate[d] * (y2[i] - xv);
}

// =====================================================================
extern "C" void kernel_launch(void* const* d_in, const int* in_sizes, int n_in,
                              void* d_out, int out_size, void* d_ws, size_t ws_size,
                              hipStream_t stream) {
    (void)in_sizes; (void)n_in; (void)out_size; (void)ws_size;

    const int*   input_ids = (const int*)  d_in[0];
    const float* tok_emb   = (const float*)d_in[1];
    const float* pos_emb   = (const float*)d_in[2];
    const float* reg_init  = (const float*)d_in[3];
    const float* s4_ln_g   = (const float*)d_in[4];
    const float* s4_ln_b   = (const float*)d_in[5];
    const float* s4_wq     = (const float*)d_in[6];
    const float* s4_wk     = (const float*)d_in[7];
    const float* s4_wv     = (const float*)d_in[8];
    const float* s3_gate_w = (const float*)d_in[9];
    const float* s3_gate_b = (const float*)d_in[10];
    const float* s3_write_w= (const float*)d_in[11];
    const float* s3_wg_w   = (const float*)d_in[12];
    const float* s3_wg_b   = (const float*)d_in[13];
    const float* ln1_g     = (const float*)d_in[14];
    const float* ln1_b     = (const float*)d_in[15];
    const float* qkvo_w    = (const float*)d_in[16];
    const float* qkvo_b    = (const float*)d_in[17];
    const float* ln2_g     = (const float*)d_in[18];
    const float* ln2_b     = (const float*)d_in[19];
    const float* ff1_w     = (const float*)d_in[20];
    const float* ff1_b     = (const float*)d_in[21];
    const float* ff2_w     = (const float*)d_in[22];
    const float* ff2_b     = (const float*)d_in[23];
    const float* out_ln_g  = (const float*)d_in[24];
    const float* out_ln_b  = (const float*)d_in[25];
    float* logits = (float*)d_out;

    // ---- bump allocator over d_ws (total ~185 MB) ----
    char*  base = (char*)d_ws;
    size_t off  = 0;
    auto alloc = [&](size_t bytes) -> char* {
        char* p = base + off;
        off += (bytes + 255) & ~(size_t)255;
        return p;
    };
    _Float16* TOK16   = (_Float16*)alloc((size_t)cV * cD * 2);
    _Float16* QKVO16  = (_Float16*)alloc((size_t)3 * 4 * cD * cD * 2);
    _Float16* FF1W16  = (_Float16*)alloc((size_t)3 * cDFF * cD * 2);
    _Float16* FF2W16  = (_Float16*)alloc((size_t)3 * cD * cDFF * 2);
    _Float16* WK16    = (_Float16*)alloc((size_t)cD * cD * 2);
    _Float16* WV16    = (_Float16*)alloc((size_t)cD * cD * 2);
    float*    X       = (float*)   alloc((size_t)cBL * cD * 4);
    _Float16* H16     = (_Float16*)alloc((size_t)cBL * cD * 2);
    float*    Q32     = (float*)   alloc((size_t)cBL * cD * 4);
    float*    K32     = (float*)   alloc((size_t)cBL * cD * 4);
    float*    V32     = (float*)   alloc((size_t)cBL * cD * 4);
    float*    O32     = (float*)   alloc((size_t)cBL * cD * 4);
    _Float16* O16     = (_Float16*)alloc((size_t)cBL * cD * 2);
    float*    Y32     = (float*)   alloc((size_t)cBL * cD * 4);
    _Float16* H2_16   = (_Float16*)alloc((size_t)cBL * cD * 2);
    float*    FF32    = (float*)   alloc((size_t)cBL * cDFF * 4);
    _Float16* FF16    = (_Float16*)alloc((size_t)cBL * cDFF * 2);
    float*    Y2_32   = (float*)   alloc((size_t)cBL * cD * 4);
    float*    REG     = (float*)   alloc(cD * 4);
    float*    QV      = (float*)   alloc(cD * 4);
    float*    SC      = (float*)   alloc(cBL * 4);
    float*    SM      = (float*)   alloc(cD * 4);
    float*    GATE    = (float*)   alloc(cD * 4);

    auto cvt = [&](const float* s, _Float16* d, size_t n) {
        k_cvt_f16<<<(unsigned)((n + 255) / 256), 256, 0, stream>>>(s, d, n);
    };
    auto gemm = [&](const _Float16* A, const _Float16* Bw, const float* bias,
                    const float* addsrc, float* C, int M, int N, int K, int act) {
        dim3 grid((N + 63) / 64, (M + 127) / 128);
        k_gemm_wmma<<<grid, 256, 0, stream>>>(A, Bw, bias, addsrc, C, M, N, K, act);
    };

    // ---- one-time f16 weight staging (tok_emb f16 = 51 MB -> fits L2) ----
    cvt(tok_emb, TOK16, (size_t)cV * cD);
    cvt(qkvo_w,  QKVO16, (size_t)3 * 4 * cD * cD);
    cvt(ff1_w,   FF1W16, (size_t)3 * cDFF * cD);
    cvt(ff2_w,   FF2W16, (size_t)3 * cD * cDFF);
    cvt(s4_wk,   WK16,   (size_t)cD * cD);
    cvt(s4_wv,   WV16,   (size_t)cD * cD);

    // ---- embedding + register attention ----
    k_copy_f32<<<2, 256, 0, stream>>>(reg_init, REG, cD);
    k_embed<<<(cBL * cD + 255) / 256, 256, 0, stream>>>(input_ids, tok_emb, pos_emb, X);
    k_ln<<<cBL / 8, 256, 0, stream>>>(X, s4_ln_g, s4_ln_b, H16, cBL);
    gemm(H16, WK16, nullptr, nullptr, K32, cBL, cD, cD, 0);
    gemm(H16, WV16, nullptr, nullptr, V32, cBL, cD, cD, 0);
    k_regq<<<1, cD, 0, stream>>>(s4_wq, REG, QV);
    k_regscore<<<cBL / 256, 256, 0, stream>>>(QV, K32, SC);
    k_regupd<<<1, cD, 0, stream>>>(SC, V32, REG);

    // ---- compressor loop ----
    const int strides[3] = {1, 8, 64};
    for (int it = 0; it < cNIT; ++it) {
        for (int i = 0; i < 3; ++i) {
            int stride = strides[i];
            const _Float16* Wq = QKVO16 + (size_t)(i * 4 + 0) * cD * cD;
            const _Float16* Wk = QKVO16 + (size_t)(i * 4 + 1) * cD * cD;
            const _Float16* Wv = QKVO16 + (size_t)(i * 4 + 2) * cD * cD;
            const _Float16* Wo = QKVO16 + (size_t)(i * 4 + 3) * cD * cD;
            const float* bq = qkvo_b + (size_t)(i * 4 + 0) * cD;
            const float* bk = qkvo_b + (size_t)(i * 4 + 1) * cD;
            const float* bv = qkvo_b + (size_t)(i * 4 + 2) * cD;
            const float* bo = qkvo_b + (size_t)(i * 4 + 3) * cD;

            k_ln<<<cBL / 8, 256, 0, stream>>>(X, ln1_g + i * cD, ln1_b + i * cD, H16, cBL);
            gemm(H16, Wq, bq, nullptr, Q32, cBL, cD, cD, 0);
            gemm(H16, Wk, bk, nullptr, K32, cBL, cD, cD, 0);
            gemm(H16, Wv, bv, nullptr, V32, cBL, cD, cD, 0);
            k_swattn<<<(cBL * cH) / 256, 256, 0, stream>>>(Q32, K32, V32, O32, stride);
            cvt(O32, O16, (size_t)cBL * cD);
            gemm(O16, Wo, bo, X, Y32, cBL, cD, cD, 0);           // y = x + o@Wo^T + b
            k_ln<<<cBL / 8, 256, 0, stream>>>(Y32, ln2_g + i * cD, ln2_b + i * cD, H2_16, cBL);
            gemm(H2_16, FF1W16 + (size_t)i * cDFF * cD, ff1_b + (size_t)i * cDFF,
                 nullptr, FF32, cBL, cDFF, cD, 1);               // gelu
            cvt(FF32, FF16, (size_t)cBL * cDFF);
            gemm(FF16, FF2W16 + (size_t)i * cD * cDFF, ff2_b + (size_t)i * cD,
                 Y32, Y2_32, cBL, cD, cDFF, 0);                  // y2 = y + ff@W2^T + b
            k_colmean<<<cD / 256, 256, 0, stream>>>(Y2_32, X, SM);
            k_gate<<<1, cD, 0, stream>>>(s3_gate_w + (size_t)i * cD * 2 * cD,
                                         s3_gate_b + (size_t)i * cD,
                                         s3_wg_w, s3_wg_b, s3_write_w, SM, REG, GATE);
            k_xupd<<<(cBL * cD) / 256, 256, 0, stream>>>(X, Y2_32, GATE);
        }
    }

    // ---- final LN + logits GEMM (A: 4096x512 f16, B: tok_emb f16 in L2) ----
    k_ln<<<cBL / 8, 256, 0, stream>>>(X, out_ln_g, out_ln_b, H16, cBL);
    gemm(H16, TOK16, nullptr, nullptr, logits, cBL, cV, cD, 0);
}
